// LigRecEGNN_80977313399015
// MI455X (gfx1250) — matrix operand order repaired and checked by reading, not compiled, not run on hardware
//
#include <hip/hip_runtime.h>
#include <hip/hip_bf16.h>

typedef __attribute__((ext_vector_type(16))) _Float16 v16h;
typedef __attribute__((ext_vector_type(8)))  float    v8f;

__device__ __forceinline__ float silu_f(float x) {
  return x / (1.0f + __expf(-x));
}

// A fragment (16x32 f16) from a row-major f16 tile (row stride 128 halves).
// ISA 7.12.2: lane m=L&15; halves 0..7 -> K = hi*8 + i, halves 8..15 -> K = 16 + hi*8 + i.
__device__ __forceinline__ v16h load_a_frag(const _Float16* __restrict__ Atile,
                                            int lane, int kt) {
  const int m = lane & 15, hi = lane >> 4;
  const _Float16* p = Atile + m * 128 + kt * 32 + hi * 8;
  v16h a;
#pragma unroll
  for (int i = 0; i < 8; ++i) { a[i] = p[i]; a[8 + i] = p[16 + i]; }
  return a;
}

// B fragment (32x16 f16) from an N-major (transposed) f16 tile Bt[n*128 + k].
// Lane: n = nt*16 + (L&15); halves 0..7 -> K = kt*32 + (L>>4)*8 + i, halves 8..15 -> K = kt*32 + 16 + (L>>4)*8 + i.
__device__ __forceinline__ v16h load_b_frag(const _Float16* __restrict__ Bt,
                                            int lane, int nt, int kt) {
  const int n = nt * 16 + (lane & 15);
  const int k0 = kt * 32 + (lane >> 4) * 8;
  const _Float16* p = Bt + n * 128 + k0;
  v16h b;
#pragma unroll
  for (int i = 0; i < 8; ++i) { b[i] = p[i]; b[8 + i] = p[16 + i]; }
  return b;
}

// C = act( A0[M,128] @ B0[128,128] (+ A1 @ B1) + bias ) (+ resid); f32 or f16 output.
__global__ __launch_bounds__(256)
void gemm128_kernel(const float* __restrict__ A0, const float* __restrict__ B0,
                    const float* __restrict__ A1, const float* __restrict__ B1,
                    const float* __restrict__ bias, const float* __restrict__ resid,
                    void* __restrict__ Cout, int M, int act_silu, int store_f16)
{
  __shared__ _Float16 Bt[128 * 128];  // 32 KB, N-major weights
  __shared__ _Float16 At[128 * 128];  // 32 KB, row-major A tile
  const int t = threadIdx.x, lane = t & 31, wave = t >> 5;
  const int rb = blockIdx.x * 128;

  for (int idx = t; idx < 128 * 128; idx += 256) {
    int k = idx >> 7, n = idx & 127;
    Bt[n * 128 + k] = (_Float16)B0[idx];
  }
  for (int idx = t; idx < 128 * 128; idx += 256) {
    int r = idx >> 7, c = idx & 127;
    int gr = rb + r;
    At[idx] = (_Float16)(gr < M ? A0[(size_t)gr * 128 + c] : 0.0f);
  }
  __syncthreads();

  v8f acc[8];
#pragma unroll
  for (int nt = 0; nt < 8; ++nt)
#pragma unroll
    for (int i = 0; i < 8; ++i) acc[nt][i] = 0.0f;

  const _Float16* Arow = At + wave * 16 * 128;
#pragma unroll
  for (int kt = 0; kt < 4; ++kt) {
    v16h a = load_a_frag(Arow, lane, kt);
#pragma unroll
    for (int nt = 0; nt < 8; ++nt) {
      v16h b = load_b_frag(Bt, lane, nt, kt);
      acc[nt] = __builtin_amdgcn_wmma_f32_16x16x32_f16(false, a, false, b,
                                                       (short)0, acc[nt], false, false);
    }
  }

  if (A1 != nullptr) {  // second K-block (for logical K=256 concat inputs)
    __syncthreads();
    for (int idx = t; idx < 128 * 128; idx += 256) {
      int k = idx >> 7, n = idx & 127;
      Bt[n * 128 + k] = (_Float16)B1[idx];
    }
    for (int idx = t; idx < 128 * 128; idx += 256) {
      int r = idx >> 7, c = idx & 127;
      int gr = rb + r;
      At[idx] = (_Float16)(gr < M ? A1[(size_t)gr * 128 + c] : 0.0f);
    }
    __syncthreads();
#pragma unroll
    for (int kt = 0; kt < 4; ++kt) {
      v16h a = load_a_frag(Arow, lane, kt);
#pragma unroll
      for (int nt = 0; nt < 8; ++nt) {
        v16h b = load_b_frag(Bt, lane, nt, kt);
        acc[nt] = __builtin_amdgcn_wmma_f32_16x16x32_f16(false, a, false, b,
                                                         (short)0, acc[nt], false, false);
      }
    }
  }

  const int hi = lane >> 4, ln = lane & 15;
#pragma unroll
  for (int nt = 0; nt < 8; ++nt) {
    int col = nt * 16 + ln;
    float bv = bias ? bias[col] : 0.0f;
#pragma unroll
    for (int r = 0; r < 8; ++r) {
      int gr = rb + wave * 16 + r + 8 * hi;
      if (gr < M) {
        float v = acc[nt][r] + bv;
        if (act_silu) v = silu_f(v);
        if (resid) v += resid[(size_t)gr * 128 + col];
        if (store_f16) ((_Float16*)Cout)[(size_t)gr * 128 + col] = (_Float16)v;
        else           ((float*)Cout)[(size_t)gr * 128 + col] = v;
      }
    }
  }
}

// Fused per-edge kernel: builds silu(P_s[src]+P_d[dst]+dij*w_last) tiles (edge MLP
// hidden), fuses the coord-MLP dot product, runs WMMA for hidden @ We2, applies
// bias+silu, and scatter-adds msg_h / coef*x_dir with global f32 atomics.
__global__ __launch_bounds__(256)
void edge_kernel(const int* __restrict__ src, const int* __restrict__ dst, int E,
                 const float* __restrict__ xs, const float* __restrict__ xd,
                 const _Float16* __restrict__ Ps, const _Float16* __restrict__ Pd,
                 const _Float16* __restrict__ Qs, const _Float16* __restrict__ Qd,
                 const float* __restrict__ wle_g, const float* __restrict__ wlc_g,
                 const float* __restrict__ We2, const float* __restrict__ be2,
                 const float* __restrict__ Wc2, const float* __restrict__ bc2,
                 float* __restrict__ h_neigh, float* __restrict__ x_neigh)
{
  __shared__ _Float16 Wt[128 * 128];  // We2 N-major, 32 KB
  __shared__ _Float16 Ae[128 * 128];  // 128 edges x 128 hidden, 32 KB
  __shared__ float wle[128], wlc[128], wc2s[128], be2s[128];
  __shared__ int sSrc[128], sDst[128];
  __shared__ float sDij[128], sDirX[128], sDirY[128], sDirZ[128];
  __shared__ float cpart[256];

  const int t = threadIdx.x, lane = t & 31, wave = t >> 5;
  const long long eb = (long long)blockIdx.x * 128;

  for (int idx = t; idx < 128 * 128; idx += 256) {
    int k = idx >> 7, n = idx & 127;
    Wt[n * 128 + k] = (_Float16)We2[idx];
  }
  if (t < 128) {
    wle[t] = wle_g[t]; wlc[t] = wlc_g[t]; wc2s[t] = Wc2[t]; be2s[t] = be2[t];
    long long e = eb + t;
    bool valid = e < (long long)E;
    int s = valid ? src[e] : 0;
    int d = valid ? dst[e] : 0;
    float dx = xs[3 * s + 0] - xd[3 * d + 0];
    float dy = xs[3 * s + 1] - xd[3 * d + 1];
    float dz = xs[3 * s + 2] - xd[3 * d + 2];
    float dij = sqrtf(dx * dx + dy * dy + dz * dz);
    float inv = 1.0f / (dij + 1e-9f);
    sSrc[t] = s; sDst[t] = valid ? d : -1;
    sDij[t] = dij;
    sDirX[t] = dx * inv; sDirY[t] = dy * inv; sDirZ[t] = dz * inv;
  }
  __syncthreads();

  {  // two threads per edge; each handles 64 of the 128 hidden channels
    const int el = t >> 1, half = t & 1, k0 = half * 64;
    const int s = sSrc[el];
    const int dr = sDst[el];
    const int d = dr < 0 ? 0 : dr;
    const float dij = sDij[el];
    const _Float16* ps = Ps + (size_t)s * 128 + k0;
    const _Float16* pd = Pd + (size_t)d * 128 + k0;
    const _Float16* qs = Qs + (size_t)s * 128 + k0;
    const _Float16* qd = Qd + (size_t)d * 128 + k0;
    _Float16* ae = Ae + el * 128 + k0;
    float part = 0.0f;
#pragma unroll 8
    for (int k = 0; k < 64; ++k) {
      int kk = k0 + k;
      float te = (float)ps[k] + (float)pd[k] + dij * wle[kk];
      ae[k] = (_Float16)silu_f(te);
      float tc = (float)qs[k] + (float)qd[k] + dij * wlc[kk];
      part += silu_f(tc) * wc2s[kk];
    }
    cpart[t] = part;
  }
  __syncthreads();

  v8f acc[8];
#pragma unroll
  for (int nt = 0; nt < 8; ++nt)
#pragma unroll
    for (int i = 0; i < 8; ++i) acc[nt][i] = 0.0f;

  const _Float16* Arow = Ae + wave * 16 * 128;
#pragma unroll
  for (int kt = 0; kt < 4; ++kt) {
    v16h a = load_a_frag(Arow, lane, kt);
#pragma unroll
    for (int nt = 0; nt < 8; ++nt) {
      v16h b = load_b_frag(Wt, lane, nt, kt);
      acc[nt] = __builtin_amdgcn_wmma_f32_16x16x32_f16(false, a, false, b,
                                                       (short)0, acc[nt], false, false);
    }
  }

  if (t < 128) {  // coord message scatter
    int d = sDst[t];
    if (d >= 0) {
      float coef = silu_f(cpart[2 * t] + cpart[2 * t + 1] + bc2[0]);
      atomicAdd(&x_neigh[3 * d + 0], coef * sDirX[t]);
      atomicAdd(&x_neigh[3 * d + 1], coef * sDirY[t]);
      atomicAdd(&x_neigh[3 * d + 2], coef * sDirZ[t]);
    }
  }

  const int hi = lane >> 4, ln = lane & 15;
#pragma unroll
  for (int nt = 0; nt < 8; ++nt) {
    int col = nt * 16 + ln;
    float bv = be2s[col];
#pragma unroll
    for (int r = 0; r < 8; ++r) {
      int el = wave * 16 + r + 8 * hi;
      int d = sDst[el];
      if (d >= 0) {
        float v = silu_f(acc[nt][r] + bv);
        atomicAdd(&h_neigh[(size_t)d * 128 + col], v);
      }
    }
  }
}

__global__ void axpy_kernel(float* __restrict__ x, const float* __restrict__ y, int n) {
  int i = blockIdx.x * 256 + threadIdx.x;
  if (i < n) x[i] += y[i];
}

extern "C" void kernel_launch(void* const* d_in, const int* in_sizes, int n_in,
                              void* d_out, int out_size, void* d_ws, size_t ws_size,
                              hipStream_t stream) {
  const float* h_lig = (const float*)d_in[0];
  const float* h_rec = (const float*)d_in[1];
  const float* x_lig = (const float*)d_in[2];
  const float* x_rec = (const float*)d_in[3];
  const int* ll_src = (const int*)d_in[4];
  const int* ll_dst = (const int*)d_in[5];
  const int* rl_src = (const int*)d_in[6];
  const int* rl_dst = (const int*)d_in[7];
  const float* We1 = (const float*)d_in[8];
  const float* be1 = (const float*)d_in[9];
  const float* We2 = (const float*)d_in[10];
  const float* be2 = (const float*)d_in[11];
  const float* Wc1 = (const float*)d_in[12];
  const float* bc1 = (const float*)d_in[13];
  const float* Wc2 = (const float*)d_in[14];
  const float* bc2 = (const float*)d_in[15];
  const float* Wn1 = (const float*)d_in[16];
  const float* bn1 = (const float*)d_in[17];
  const float* Wn2 = (const float*)d_in[18];
  const float* bn2 = (const float*)d_in[19];
  (void)n_in; (void)out_size; (void)ws_size;

  const int D = 128, H = 128;
  const int n_lig = in_sizes[0] / D;
  const int n_rec = in_sizes[1] / D;
  const int Ell = in_sizes[4];
  const int Erl = in_sizes[6];
  const int L = in_sizes[8] / (2 * 257 * H);

  char* base = (char*)d_ws;
  size_t off = 0;
  auto walloc = [&](size_t bytes) -> void* {
    void* p = (void*)(base + off);
    off = (off + bytes + 255) & ~(size_t)255;
    return p;
  };
  float* hlA = (float*)walloc((size_t)n_lig * D * 4);
  float* hlB = (float*)walloc((size_t)n_lig * D * 4);
  float* z1  = (float*)walloc((size_t)n_lig * H * 4);
  float* hN  = (float*)walloc((size_t)n_lig * H * 4);
  float* xN  = (float*)walloc((size_t)n_lig * 3 * 4);
  float* xl  = (float*)walloc((size_t)n_lig * 3 * 4);
  _Float16* Plls = (_Float16*)walloc((size_t)n_lig * H * 2);
  _Float16* Plld = (_Float16*)walloc((size_t)n_lig * H * 2);
  _Float16* Qlls = (_Float16*)walloc((size_t)n_lig * H * 2);
  _Float16* Qlld = (_Float16*)walloc((size_t)n_lig * H * 2);
  _Float16* Prld = (_Float16*)walloc((size_t)n_lig * H * 2);
  _Float16* Qrld = (_Float16*)walloc((size_t)n_lig * H * 2);
  _Float16* Prls = (_Float16*)walloc((size_t)n_rec * H * 2);
  _Float16* Qrls = (_Float16*)walloc((size_t)n_rec * H * 2);

  hipMemcpyAsync(hlA, h_lig, (size_t)n_lig * D * 4, hipMemcpyDeviceToDevice, stream);
  hipMemcpyAsync(xl, x_lig, (size_t)n_lig * 3 * 4, hipMemcpyDeviceToDevice, stream);

  float* hcur = hlA;
  float* hnext = hlB;
  const int gbL = (n_lig + 127) / 128;
  const int gbR = (n_rec + 127) / 128;
  dim3 blk(256);

  for (int l = 0; l < L; ++l) {
    hipMemsetAsync(hN, 0, (size_t)n_lig * H * 4, stream);
    hipMemsetAsync(xN, 0, (size_t)n_lig * 3 * 4, stream);

    const float* We1_ll = We1 + (size_t)(l * 2 + 0) * 257 * H;
    const float* We1_rl = We1 + (size_t)(l * 2 + 1) * 257 * H;
    const float* Wc1_ll = Wc1 + (size_t)(l * 2 + 0) * 257 * H;
    const float* Wc1_rl = Wc1 + (size_t)(l * 2 + 1) * 257 * H;
    const float* be1_ll = be1 + (size_t)(l * 2 + 0) * H;
    const float* be1_rl = be1 + (size_t)(l * 2 + 1) * H;
    const float* bc1_ll = bc1 + (size_t)(l * 2 + 0) * H;
    const float* bc1_rl = bc1 + (size_t)(l * 2 + 1) * H;
    const float* We2_ll = We2 + (size_t)(l * 2 + 0) * H * H;
    const float* We2_rl = We2 + (size_t)(l * 2 + 1) * H * H;
    const float* be2_ll = be2 + (size_t)(l * 2 + 0) * H;
    const float* be2_rl = be2 + (size_t)(l * 2 + 1) * H;
    const float* Wc2_ll = Wc2 + (size_t)(l * 2 + 0) * H;
    const float* Wc2_rl = Wc2 + (size_t)(l * 2 + 1) * H;
    const float* bc2_ll = bc2 + (size_t)(l * 2 + 0);
    const float* bc2_rl = bc2 + (size_t)(l * 2 + 1);

    // Per-node precompute of first-layer partial products (bias folded into dst side).
    gemm128_kernel<<<gbL, blk, 0, stream>>>(hcur,  We1_ll,           nullptr, nullptr, nullptr, nullptr, Plls, n_lig, 0, 1);
    gemm128_kernel<<<gbL, blk, 0, stream>>>(hcur,  We1_ll + 128 * H, nullptr, nullptr, be1_ll, nullptr, Plld, n_lig, 0, 1);
    gemm128_kernel<<<gbL, blk, 0, stream>>>(hcur,  Wc1_ll,           nullptr, nullptr, nullptr, nullptr, Qlls, n_lig, 0, 1);
    gemm128_kernel<<<gbL, blk, 0, stream>>>(hcur,  Wc1_ll + 128 * H, nullptr, nullptr, bc1_ll, nullptr, Qlld, n_lig, 0, 1);
    gemm128_kernel<<<gbR, blk, 0, stream>>>(h_rec, We1_rl,           nullptr, nullptr, nullptr, nullptr, Prls, n_rec, 0, 1);
    gemm128_kernel<<<gbL, blk, 0, stream>>>(hcur,  We1_rl + 128 * H, nullptr, nullptr, be1_rl, nullptr, Prld, n_lig, 0, 1);
    gemm128_kernel<<<gbR, blk, 0, stream>>>(h_rec, Wc1_rl,           nullptr, nullptr, nullptr, nullptr, Qrls, n_rec, 0, 1);
    gemm128_kernel<<<gbL, blk, 0, stream>>>(hcur,  Wc1_rl + 128 * H, nullptr, nullptr, bc1_rl, nullptr, Qrld, n_lig, 0, 1);

    // Fused edge message + WMMA + scatter (both edge types into same accumulators).
    edge_kernel<<<(Ell + 127) / 128, blk, 0, stream>>>(
        ll_src, ll_dst, Ell, xl, xl, Plls, Plld, Qlls, Qlld,
        We1_ll + 256 * H, Wc1_ll + 256 * H, We2_ll, be2_ll, Wc2_ll, bc2_ll, hN, xN);
    edge_kernel<<<(Erl + 127) / 128, blk, 0, stream>>>(
        rl_src, rl_dst, Erl, x_rec, xl, Prls, Prld, Qrls, Qrld,
        We1_rl + 256 * H, Wc1_rl + 256 * H, We2_rl, be2_rl, Wc2_rl, bc2_rl, hN, xN);

    // Node MLP: z1 = silu([h, h_neigh] @ Wn1 + bn1); h' = h + z1 @ Wn2 + bn2
    const float* Wn1_l = Wn1 + (size_t)l * 256 * H;
    gemm128_kernel<<<gbL, blk, 0, stream>>>(hcur, Wn1_l, hN, Wn1_l + 128 * H,
                                            bn1 + (size_t)l * H, nullptr, z1, n_lig, 1, 0);
    gemm128_kernel<<<gbL, blk, 0, stream>>>(z1, Wn2 + (size_t)l * H * D, nullptr, nullptr,
                                            bn2 + (size_t)l * D, hcur, hnext, n_lig, 0, 0);
    axpy_kernel<<<(n_lig * 3 + 255) / 256, blk, 0, stream>>>(xl, xN, n_lig * 3);

    float* tmp = hcur; hcur = hnext; hnext = tmp;
  }

  hipMemcpyAsync(d_out, hcur, (size_t)n_lig * D * 4, hipMemcpyDeviceToDevice, stream);
  hipMemcpyAsync((float*)d_out + (size_t)n_lig * D, xl, (size_t)n_lig * 3 * 4,
                 hipMemcpyDeviceToDevice, stream);
}
